// KannadaSelfAttention_31207232373129
// MI455X (gfx1250) — matrix operand order repaired and checked by reading, not compiled, not run on hardware
//
#include <hip/hip_runtime.h>
#include <hip/hip_bf16.h>

// Problem constants (BERT-base style self-attention)
#define B_  8
#define S_  512
#define H_  768
#define NH_ 12
#define HD_ 64

typedef _Float16 half8  __attribute__((ext_vector_type(8)));
typedef _Float16 half16 __attribute__((ext_vector_type(16)));
typedef float    float8 __attribute__((ext_vector_type(8)));
typedef float    f32x4  __attribute__((ext_vector_type(4)));

// CDNA5 async global->LDS copy (CDNA5 ISA): per-lane
// 16 bytes from global vaddr to LDS address vdst, tracked by ASYNCcnt.
#define ASYNC_LOAD_B128(ldsaddr, gaddr)                         \
    asm volatile("global_load_async_to_lds_b128 %0, %1, off"    \
                 :: "v"(ldsaddr), "v"(gaddr) : "memory")
#define WAIT_ASYNC0() asm volatile("s_wait_asynccnt 0" ::: "memory")

static __device__ __forceinline__ half16 frag_from(const _Float16* p0, const _Float16* p1) {
    half8 lo = *(const half8*)p0;
    half8 hi = *(const half8*)p1;
    half16 r;
#pragma unroll
    for (int i = 0; i < 8; ++i) { r[i] = lo[i]; r[i + 8] = hi[i]; }
    return r;
}

static __device__ __forceinline__ float8 f8zero() {
    float8 z;
#pragma unroll
    for (int i = 0; i < 8; ++i) z[i] = 0.0f;
    return z;
}

// Pack two f32x4 into one half8 (register-side convert -> b128 LDS store)
static __device__ __forceinline__ half8 cvt_pack(f32x4 a, f32x4 b) {
    half8 h;
#pragma unroll
    for (int i = 0; i < 4; ++i) { h[i] = (_Float16)a[i]; h[i + 4] = (_Float16)b[i]; }
    return h;
}

// ---------------------------------------------------------------------------
// Kernel 1: Y = X @ W^T + b, written as f16 [B, NH, S, HD] into workspace.
// Block = 256 threads (8 waves). Block tile 128(M) x 128(N). Wave tile 32x64
// as 2x4 WMMA 16x16 tiles, K-loop in steps of 32, software-pipelined
// register-staged global->LDS copies (all vector b128 traffic).
// ---------------------------------------------------------------------------
__global__ __launch_bounds__(256) void qkv_proj_kernel(
    const float* __restrict__ hs,    // [4096, 768] row-major (B*S tokens)
    const float* __restrict__ W,     // [768, 768] row-major; Y = X @ W^T
    const float* __restrict__ bias,  // [768]
    _Float16* __restrict__ dst)      // [B, NH, S, HD] f16
{
    __shared__ _Float16 Xs[128 * 32];
    __shared__ _Float16 Ws[128 * 32];

    const int tid    = threadIdx.x;
    const int lane   = tid & 31;
    const int waveId = tid >> 5;     // 0..7
    const int waveM  = waveId >> 1;  // 0..3 -> 32-row strip
    const int waveN  = waveId & 1;   // 0..1 -> 64-col strip
    const int idx    = lane & 15;
    const int hid    = lane >> 4;
    const int mBase  = blockIdx.x * 128;
    const int nBase  = blockIdx.y * 128;

    // Per-thread staging slices (16 contiguous floats each for X and W):
    const int sRow = tid >> 1, sCol = (tid & 1) * 16;
    const float* xg = hs + (size_t)(mBase + sRow) * H_ + sCol;
    const float* wg = W  + (size_t)(nBase + sRow) * H_ + sCol;
    _Float16* xs = &Xs[sRow * 32 + sCol];
    _Float16* ws = &Ws[sRow * 32 + sCol];

    f32x4 xr[4], wr[4];
#pragma unroll
    for (int i = 0; i < 4; ++i) xr[i] = *(const f32x4*)(xg + i * 4);
#pragma unroll
    for (int i = 0; i < 4; ++i) wr[i] = *(const f32x4*)(wg + i * 4);

    float8 acc[2][4];
#pragma unroll
    for (int i = 0; i < 2; ++i)
#pragma unroll
        for (int j = 0; j < 4; ++j) acc[i][j] = f8zero();

    for (int k0 = 0; k0 < H_; k0 += 32) {
        __syncthreads();
        // Stage current tile regs -> LDS (vector b128 stores)
        *(half8*)(xs)     = cvt_pack(xr[0], xr[1]);
        *(half8*)(xs + 8) = cvt_pack(xr[2], xr[3]);
        *(half8*)(ws)     = cvt_pack(wr[0], wr[1]);
        *(half8*)(ws + 8) = cvt_pack(wr[2], wr[3]);
        __syncthreads();

        // Issue next tile's global loads early (overlap with WMMA below)
        if (k0 + 32 < H_) {
#pragma unroll
            for (int i = 0; i < 4; ++i) xr[i] = *(const f32x4*)(xg + k0 + 32 + i * 4);
#pragma unroll
            for (int i = 0; i < 4; ++i) wr[i] = *(const f32x4*)(wg + k0 + 32 + i * 4);
        }

        half16 a[2], bf[4];
#pragma unroll
        for (int sm = 0; sm < 2; ++sm) {
            int m  = waveM * 32 + sm * 16 + idx;
            int kb = hid * 8;
            a[sm] = frag_from(&Xs[m * 32 + kb], &Xs[m * 32 + kb + 16]);
        }
#pragma unroll
        for (int sn = 0; sn < 4; ++sn) {
            int n  = waveN * 64 + sn * 16 + idx;
            int kb = hid * 8;
            bf[sn] = frag_from(&Ws[n * 32 + kb], &Ws[n * 32 + kb + 16]);
        }
#pragma unroll
        for (int sm = 0; sm < 2; ++sm)
#pragma unroll
            for (int sn = 0; sn < 4; ++sn)
                acc[sm][sn] = __builtin_amdgcn_wmma_f32_16x16x32_f16(
                    false, a[sm], false, bf[sn], (short)0, acc[sm][sn], false, false);
    }

    // Epilogue: add bias, scatter to [B, NH, S, HD] f16
#pragma unroll
    for (int sm = 0; sm < 2; ++sm)
#pragma unroll
        for (int sn = 0; sn < 4; ++sn) {
            int nG = nBase + waveN * 64 + sn * 16 + idx;
            float bv = bias[nG];
            int nh = nG >> 6, hd = nG & 63;
#pragma unroll
            for (int r = 0; r < 8; ++r) {
                int mG = mBase + waveM * 32 + sm * 16 + r + 8 * hid;
                int bb = mG >> 9;        // token / 512
                int ss = mG & 511;
                dst[(((size_t)bb * NH_ + nh) * S_ + ss) * HD_ + hd] =
                    (_Float16)(acc[sm][sn][r] + bv);
            }
        }
}

// ---------------------------------------------------------------------------
// Kernel 2: flash-attention. Block = 128 threads (4 waves), one (b, head,
// 128-query tile). Each wave owns 32 query rows (2 WMMA M-tiles). K blocks
// are fetched with double-buffered async global->LDS copies (ASYNCcnt);
// V goes through registers (transpose at staging). Online softmax; P is
// converted to A-fragment layout via a per-wave LDS round-trip.
// ---------------------------------------------------------------------------
__global__ __launch_bounds__(128) void attention_kernel(
    const _Float16* __restrict__ q,   // [B, NH, S, HD]
    const _Float16* __restrict__ k,
    const _Float16* __restrict__ v,
    const float* __restrict__ mask,   // [B, 1, 1, S]
    float* __restrict__ out)          // [B, S, H]
{
    __shared__ _Float16 Kbuf[2][64 * 64];     // [key][hd], double-buffered
    __shared__ _Float16 Vt[64 * 64];          // transposed: [hd][key]
    __shared__ _Float16 Pt[4 * 32 * 64];      // per-wave P staging (32 rows)

    const int tid  = threadIdx.x;
    const int lane = tid & 31;
    const int wave = tid >> 5;     // 0..3
    const int idx  = lane & 15;
    const int hid  = lane >> 4;

    const int qblk = blockIdx.x;   // 0..3 (128 queries per block)
    const int nh   = blockIdx.y;
    const int b    = blockIdx.z;
    const int bh   = b * NH_ + nh;

    const _Float16* qg = q + ((size_t)bh * S_ + qblk * 128 + wave * 32) * HD_;
    const _Float16* kg = k + (size_t)bh * S_ * HD_ + tid * 32;  // staging slice
    const _Float16* vg = v + (size_t)bh * S_ * HD_ + tid * 32;

    // LDS byte addresses of this thread's K staging slice in each buffer
    uint32_t ldsK[2];
    ldsK[0] = (uint32_t)(uintptr_t)&Kbuf[0][tid * 32];
    ldsK[1] = (uint32_t)(uintptr_t)&Kbuf[1][tid * 32];

    // Q fragments: 2 M-tiles x 2 K-steps (HD contiguous -> direct b128 loads)
    half16 qa[2][2];
#pragma unroll
    for (int sm = 0; sm < 2; ++sm)
#pragma unroll
        for (int ks = 0; ks < 2; ++ks) {
            const _Float16* base = qg + (sm * 16 + idx) * HD_ + ks * 32 + hid * 8;
            qa[sm][ks] = frag_from(base, base + 16);
        }

    // Kick off async copy of K block 0 into Kbuf[0]; preload V block 0 to regs
#pragma unroll
    for (int i = 0; i < 4; ++i)
        ASYNC_LOAD_B128(ldsK[0] + i * 16, kg + i * 8);
    half8 vrg[4];
#pragma unroll
    for (int i = 0; i < 4; ++i) vrg[i] = *(const half8*)(vg + i * 8);

    float8 o[2][4];
#pragma unroll
    for (int sm = 0; sm < 2; ++sm)
#pragma unroll
        for (int t = 0; t < 4; ++t) o[sm][t] = f8zero();
    float rowm[2][8], rowl[2][8];
#pragma unroll
    for (int sm = 0; sm < 2; ++sm)
#pragma unroll
        for (int r = 0; r < 8; ++r) { rowm[sm][r] = -1e30f; rowl[sm][r] = 0.0f; }

    for (int jb = 0; jb < 8; ++jb) {
        const int cur = jb & 1;
        WAIT_ASYNC0();            // K block jb has landed in Kbuf[cur]
        __syncthreads();          // all waves' asyncs done; Vt readers done
        // Stage V block transposed Vt[hd][key]: scalar b16 stores (transpose)
#pragma unroll
        for (int i = 0; i < 4; ++i)
#pragma unroll
            for (int j = 0; j < 8; ++j) {
                int e   = tid * 32 + i * 8 + j;
                int key = e >> 6, hd = e & 63;
                Vt[hd * 64 + key] = vrg[i][j];
            }
        __syncthreads();

        // Prefetch next key-block: K via async into idle buffer, V to regs
        if (jb + 1 < 8) {
#pragma unroll
            for (int i = 0; i < 4; ++i)
                ASYNC_LOAD_B128(ldsK[1 - cur] + i * 16,
                                kg + (size_t)(jb + 1) * 64 * HD_ + i * 8);
#pragma unroll
            for (int i = 0; i < 4; ++i)
                vrg[i] = *(const half8*)(vg + (size_t)(jb + 1) * 64 * HD_ + i * 8);
        }

        const _Float16* Kc = Kbuf[cur];

        // Score B-fragments (shared across both M-tiles)
        half16 kb[4][2];
#pragma unroll
        for (int t = 0; t < 4; ++t)
#pragma unroll
            for (int ks = 0; ks < 2; ++ks) {
                const _Float16* bp = &Kc[(t * 16 + idx) * 64 + ks * 32 + hid * 8];
                kb[t][ks] = frag_from(bp, bp + 16);
            }

        // Scores: S[32 x 64] = Q (32x64) x K^T
        float8 sc[2][4];
#pragma unroll
        for (int sm = 0; sm < 2; ++sm)
#pragma unroll
            for (int t = 0; t < 4; ++t) {
                sc[sm][t] = f8zero();
#pragma unroll
                for (int ks = 0; ks < 2; ++ks)
                    sc[sm][t] = __builtin_amdgcn_wmma_f32_16x16x32_f16(
                        false, qa[sm][ks], false, kb[t][ks], (short)0,
                        sc[sm][t], false, false);
            }
#pragma unroll
        for (int t = 0; t < 4; ++t) {
            float mv = mask[b * S_ + jb * 64 + t * 16 + idx];
#pragma unroll
            for (int sm = 0; sm < 2; ++sm)
#pragma unroll
                for (int r = 0; r < 8; ++r)
                    sc[sm][t][r] = sc[sm][t][r] * 0.125f + mv;
        }

        // Online softmax per M-tile: row-max, rescale, exponentiate, row-sum
#pragma unroll
        for (int sm = 0; sm < 2; ++sm) {
            float bm[8];
#pragma unroll
            for (int r = 0; r < 8; ++r) {
                float m = sc[sm][0][r];
#pragma unroll
                for (int t = 1; t < 4; ++t) m = fmaxf(m, sc[sm][t][r]);
#pragma unroll
                for (int off = 8; off > 0; off >>= 1)
                    m = fmaxf(m, __shfl_xor(m, off, 32));  // within 16-lane half
                bm[r] = m;
            }
            float alpha[8];
#pragma unroll
            for (int r = 0; r < 8; ++r) {
                float nm = fmaxf(rowm[sm][r], bm[r]);
                alpha[r] = __expf(rowm[sm][r] - nm);
                rowm[sm][r] = nm;
            }
            float ps[8];
#pragma unroll
            for (int r = 0; r < 8; ++r) ps[r] = 0.0f;
#pragma unroll
            for (int t = 0; t < 4; ++t)
#pragma unroll
                for (int r = 0; r < 8; ++r) {
                    float p = __expf(sc[sm][t][r] - rowm[sm][r]);
                    sc[sm][t][r] = p;
                    ps[r] += p;
                }
#pragma unroll
            for (int r = 0; r < 8; ++r) {
#pragma unroll
                for (int off = 8; off > 0; off >>= 1)
                    ps[r] += __shfl_xor(ps[r], off, 32);
                rowl[sm][r] = rowl[sm][r] * alpha[r] + ps[r];
            }
#pragma unroll
            for (int t = 0; t < 4; ++t)
#pragma unroll
                for (int r = 0; r < 8; ++r) o[sm][t][r] *= alpha[r];

            // Stage P tile (C/D layout -> [m][key] f16, per-wave LDS region)
#pragma unroll
            for (int t = 0; t < 4; ++t)
#pragma unroll
                for (int r = 0; r < 8; ++r)
                    Pt[wave * 2048 + (sm * 16 + r + 8 * hid) * 64 + t * 16 + idx] =
                        (_Float16)sc[sm][t][r];
        }
        __syncthreads();

        // O += P (32x64) x V (64x64); V transposed so B-frags are contiguous
        half16 vb[4][2];
#pragma unroll
        for (int t = 0; t < 4; ++t)
#pragma unroll
            for (int ks = 0; ks < 2; ++ks) {
                const _Float16* bp = &Vt[(t * 16 + idx) * 64 + ks * 32 + hid * 8];
                vb[t][ks] = frag_from(bp, bp + 16);
            }
        half16 pa[2][2];
#pragma unroll
        for (int sm = 0; sm < 2; ++sm)
#pragma unroll
            for (int ks = 0; ks < 2; ++ks) {
                const _Float16* p =
                    &Pt[wave * 2048 + (sm * 16 + idx) * 64 + ks * 32 + hid * 8];
                pa[sm][ks] = frag_from(p, p + 16);
            }
#pragma unroll
        for (int sm = 0; sm < 2; ++sm)
#pragma unroll
            for (int t = 0; t < 4; ++t)
#pragma unroll
                for (int ks = 0; ks < 2; ++ks)
                    o[sm][t] = __builtin_amdgcn_wmma_f32_16x16x32_f16(
                        false, pa[sm][ks], false, vb[t][ks], (short)0,
                        o[sm][t], false, false);
    }

    // Epilogue: divide by row sum, write [B, S, H] fp32
#pragma unroll
    for (int sm = 0; sm < 2; ++sm)
#pragma unroll
        for (int t = 0; t < 4; ++t) {
            int col = nh * 64 + t * 16 + idx;
#pragma unroll
            for (int r = 0; r < 8; ++r) {
                int s = qblk * 128 + wave * 32 + sm * 16 + r + 8 * hid;
                out[((size_t)b * S_ + s) * H_ + col] = o[sm][t][r] / rowl[sm][r];
            }
        }
}

// ---------------------------------------------------------------------------
extern "C" void kernel_launch(void* const* d_in, const int* in_sizes, int n_in,
                              void* d_out, int out_size, void* d_ws, size_t ws_size,
                              hipStream_t stream) {
    const float* hs   = (const float*)d_in[0];
    const float* mask = (const float*)d_in[1];
    const float* Wq   = (const float*)d_in[2];
    const float* bq   = (const float*)d_in[3];
    const float* Wk   = (const float*)d_in[4];
    const float* bk   = (const float*)d_in[5];
    const float* Wv   = (const float*)d_in[6];
    const float* bv   = (const float*)d_in[7];

    _Float16* qkv = (_Float16*)d_ws;              // 3 x [B,NH,S,HD] f16 ~ 19 MB
    const size_t per = (size_t)B_ * NH_ * S_ * HD_;

    dim3 g1(B_ * S_ / 128, H_ / 128);
    dim3 blk1(256);
    qkv_proj_kernel<<<g1, blk1, 0, stream>>>(hs, Wq, bq, qkv);
    qkv_proj_kernel<<<g1, blk1, 0, stream>>>(hs, Wk, bk, qkv + per);
    qkv_proj_kernel<<<g1, blk1, 0, stream>>>(hs, Wv, bv, qkv + 2 * per);

    dim3 g2(S_ / 128, NH_, B_);
    dim3 blk2(128);
    attention_kernel<<<g2, blk2, 0, stream>>>(qkv, qkv + per, qkv + 2 * per,
                                              mask, (float*)d_out);
}
